// SelfAttention_40621800686059
// MI455X (gfx1250) — matrix operand order repaired and checked by reading, not compiled, not run on hardware
//
#include <hip/hip_runtime.h>

// Problem constants
#define BB    2
#define SS    2048
#define HH    1024
#define NH    16
#define HD    64
#define BS    (BB * SS)      // 4096 rows
#define N3    (3 * HH)       // 3072 qkv cols

typedef __attribute__((ext_vector_type(2)))  __bf16 bf16x2;
typedef __attribute__((ext_vector_type(16))) __bf16 v16bf;
typedef __attribute__((ext_vector_type(8)))  float  v8f;

typedef __attribute__((ext_vector_type(4))) unsigned tdm_u32x4;
typedef __attribute__((ext_vector_type(8))) int      tdm_i32x8;
typedef __attribute__((ext_vector_type(4))) int      tdm_i32x4;

#if __has_builtin(__builtin_amdgcn_tensor_load_to_lds)
#define USE_TDM 1
#else
#define USE_TDM 0
#endif

// Native bf16 convert (RNE) -> v_cvt_pk_bf16_f32 path instead of integer bit tricks
static __device__ __forceinline__ __bf16 f2bf(float f) { return (__bf16)f; }

// A fragment (16x32 bf16): lane<16 -> row=lane, K={0..15}; lane>=16 -> K={16..31}
static __device__ __forceinline__ v16bf load_frag_a(const __bf16* p0, int stride, int lane) {
    const int row = lane & 15;
    const int ko  = (lane >> 4) << 4;       // 0 or 16
    const __bf16* p = p0 + row * stride + ko;
    v16bf a;
#pragma unroll
    for (int i = 0; i < 4; ++i) {
        bf16x2 x = *(const bf16x2*)(p + 2 * i);
        bf16x2 y = *(const bf16x2*)(p + 8 + 2 * i);
        a[2 * i]     = x.x; a[2 * i + 1] = x.y;
        a[8 + 2 * i] = y.x; a[9 + 2 * i] = y.y;
    }
    return a;
}

// B fragment (32x16 bf16), source k-major: row n = output col, contiguous K pairs.
static __device__ __forceinline__ v16bf load_frag_b(const __bf16* p0, int stride, int lane, int n0) {
    const int n  = (lane & 15) + n0;
    const int ko = (lane >> 4) << 4;        // 0 or 16
    const __bf16* p = p0 + n * stride + ko;
    v16bf b;
#pragma unroll
    for (int i = 0; i < 8; ++i) {
        bf16x2 x = *(const bf16x2*)(p + 2 * i);
        b[2 * i] = x.x; b[2 * i + 1] = x.y;
    }
    return b;
}

// B fragment from row-major [K rows][N cols] tile (K pairs strided by row pitch)
static __device__ __forceinline__ v16bf load_frag_b_rowmajor(const __bf16* p0, int stride, int lane, int n0) {
    const int n  = (lane & 15) + n0;        // output column
    const int kk = (lane >> 4) << 4;        // 0 or 16 (K row offset)
    const __bf16* p = p0 + kk * stride + n;
    v16bf b;
#pragma unroll
    for (int i = 0; i < 8; ++i) {
        b[2 * i]     = p[(2 * i) * stride];
        b[2 * i + 1] = p[(2 * i + 1) * stride];
    }
    return b;
}

static __device__ __forceinline__ v8f wmma_bf16(v16bf a, v16bf b, v8f c) {
    return __builtin_amdgcn_wmma_f32_16x16x32_bf16(false, a, false, b, (short)0, c, false, false);
}

#if USE_TDM
// Issue a TDM 2D tile load: bf16 tensor [tensor_d1][tensor_d0] (row pitch stride0 elems),
// tile tile_d1 x tile_d0 -> LDS at lds_off with +1 DWORD pad per 32 DWORDs (row stride 66 elems).
static __device__ __forceinline__ void tdm_load_tile_2d_bf16(
    unsigned lds_off, const void* gaddr,
    unsigned tensor_d0, unsigned tensor_d1, unsigned stride0,
    unsigned tile_d0, unsigned tile_d1)
{
    const unsigned long long ga = (unsigned long long)(size_t)gaddr;
    tdm_u32x4 g0;
    g0[0] = 1u;                                            // count=1 (valid), user mode
    g0[1] = lds_off;                                       // LDS byte address
    g0[2] = (unsigned)(ga & 0xFFFFFFFFu);                  // global_addr[31:0]
    g0[3] = (unsigned)((ga >> 32) & 0x01FFFFFFu) | (2u << 30);  // global_addr[56:32] | type=2

    tdm_i32x8 g1;
    // data_size=1 (2B) | pad_enable | pad_interval=4 (32 DWORDs) | pad_amount=0 (1 DWORD)
    g1[0] = (int)((1u << 16) | (1u << 20) | (4u << 22));
    g1[1] = (int)(tensor_d0 << 16);                        // tensor_dim0[15:0] @63:48
    g1[2] = (int)((tensor_d0 >> 16) | (tensor_d1 << 16));  // dim0 hi | dim1 lo
    g1[3] = (int)((tensor_d1 >> 16) | (tile_d0 << 16));    // dim1 hi | tile_dim0
    g1[4] = (int)(tile_d1 & 0xFFFFu);                      // tile_dim1 (tile_dim2 = 0)
    g1[5] = (int)stride0;                                  // tensor_dim0_stride[31:0]
    g1[6] = 0;
    g1[7] = 0;

    tdm_i32x4 z4 = {0, 0, 0, 0};
#if defined(__clang_major__) && (__clang_major__ >= 23)
    tdm_i32x8 z8 = {0, 0, 0, 0, 0, 0, 0, 0};
    __builtin_amdgcn_tensor_load_to_lds(g0, g1, z4, z4, z8, 0);
#else
    __builtin_amdgcn_tensor_load_to_lds(g0, g1, z4, z4, 0);
#endif
}
#endif // USE_TDM

// ---------------------------------------------------------------------------
// Kernel 1: qkv = x @ w_qkv + b_qkv ; scatter to Q (x 1/8), K, V bf16 [B,NH,S,HD]
// ---------------------------------------------------------------------------
__global__ __launch_bounds__(256) void qkv_gemm_kernel(
    const float* __restrict__ X,     // [BS][HH]
    const float* __restrict__ W,     // [HH][N3]
    const float* __restrict__ bias,  // [N3]
    __bf16* __restrict__ Qb, __bf16* __restrict__ Kb, __bf16* __restrict__ Vb)
{
    __shared__ __bf16 As[64][40];
    __shared__ __bf16 BsT[128][40];  // transposed: [n][k]

    const int tid  = threadIdx.x;
    const int lane = tid & 31;
    const int wid  = tid >> 5;
    const int wm   = wid & 3;
    const int wn   = wid >> 2;
    const int m0   = blockIdx.y * 64;
    const int n0   = blockIdx.x * 128;

    v8f acc[4] = {};

    for (int k0 = 0; k0 < HH; k0 += 32) {
        // Stage A: 64x32 fp32 -> bf16
#pragma unroll
        for (int j = 0; j < 2; ++j) {
            int idx = tid * 2 + j;
            int row = idx >> 3;
            int c4  = (idx & 7) << 2;
            float4 f = *(const float4*)(X + (size_t)(m0 + row) * HH + k0 + c4);
            As[row][c4 + 0] = f2bf(f.x);
            As[row][c4 + 1] = f2bf(f.y);
            As[row][c4 + 2] = f2bf(f.z);
            As[row][c4 + 3] = f2bf(f.w);
        }
        // Stage B transposed: 32x128 fp32 -> bf16
#pragma unroll
        for (int j = 0; j < 4; ++j) {
            int idx = tid * 4 + j;
            int row = idx >> 5;
            int c4  = (idx & 31) << 2;
            float4 f = *(const float4*)(W + (size_t)(k0 + row) * N3 + n0 + c4);
            BsT[c4 + 0][row] = f2bf(f.x);
            BsT[c4 + 1][row] = f2bf(f.y);
            BsT[c4 + 2][row] = f2bf(f.z);
            BsT[c4 + 3][row] = f2bf(f.w);
        }
        if (k0 + 32 < HH) {  // hint next tiles toward L2
            __builtin_prefetch(X + (size_t)(m0 + (tid >> 2)) * HH + k0 + 32, 0, 1);
            __builtin_prefetch(W + (size_t)(k0 + 32 + (tid >> 3)) * N3 + n0, 0, 1);
        }
        __syncthreads();

        v16bf a = load_frag_a(&As[wm * 16][0], 40, lane);
#pragma unroll
        for (int t = 0; t < 4; ++t) {
            v16bf b = load_frag_b(&BsT[0][0], 40, lane, wn * 64 + t * 16);
            acc[t] = wmma_bf16(a, b, acc[t]);
        }
        __syncthreads();
    }

    const int rowlo = (lane >> 4) << 3;
    const int col   = lane & 15;
#pragma unroll
    for (int t = 0; t < 4; ++t) {
        const int n     = n0 + wn * 64 + t * 16 + col;
        const int which = n >> 10;           // 0=Q 1=K 2=V
        const int head  = (n & 1023) >> 6;
        const int d     = n & 63;
        __bf16* dst = (which == 0) ? Qb : ((which == 1) ? Kb : Vb);
        const float sc = (which == 0) ? 0.125f : 1.0f;   // 1/sqrt(64) folded into Q
        const float bv = bias[n];
#pragma unroll
        for (int r = 0; r < 8; ++r) {
            const int m  = m0 + wm * 16 + rowlo + r;
            const int bi = m >> 11;
            const int si = m & 2047;
            dst[(((size_t)bi * NH + head) * SS + si) * HD + d] = f2bf((acc[t][r] + bv) * sc);
        }
    }
}

// ---------------------------------------------------------------------------
// Kernel 2: flash attention. Block = (b, head, 64-row q tile), 128 threads = 4 waves.
// K/V tiles streamed by the Tensor Data Mover, double-buffered on TENSORcnt.
// ---------------------------------------------------------------------------
// Manual LDS layout (single shared object => base offset 0):
#define AT_QS_OFF   0u                         // 64 x 72 bf16 = 9216 B
#define AT_KS_OFF(i) (9216u  + (i) * 4224u)    // 2 x (32 x 66 bf16)
#define AT_VS_OFF(i) (17664u + (i) * 4224u)    // 2 x (32 x 66 bf16)
#define AT_PS_OFF   26112u                     // 4 x 16 x 40 bf16 = 5120 B
#define AT_SMEM_BYTES 31232u

__global__ __launch_bounds__(128) void attn_kernel(
    const __bf16* __restrict__ Qb, const __bf16* __restrict__ Kb,
    const __bf16* __restrict__ Vb, __bf16* __restrict__ AOb)  // AOb: [BS][HH] bf16
{
    __shared__ __align__(16) char smem[AT_SMEM_BYTES];
    __bf16* const Qs = (__bf16*)(smem + AT_QS_OFF);          // stride 72
    __bf16* const Pw0 = (__bf16*)(smem + AT_PS_OFF);         // stride 40, per wave

    const int qt  = blockIdx.x;
    const int h   = blockIdx.y;
    const int bz  = blockIdx.z;
    const int tid = threadIdx.x, lane = tid & 31, wid = tid >> 5;

    const size_t bh = ((size_t)bz * NH + h) * SS * HD;
    const int q0 = qt * 64;

    // Load Q tile 64x64 (512 uint4, 4 per thread); row stride 72 -> 144B (16B aligned)
#pragma unroll
    for (int j = 0; j < 4; ++j) {
        int idx = tid * 4 + j;
        int row = idx >> 3;
        int c8  = (idx & 7) * 8;
        *(uint4*)(Qs + row * 72 + c8) = *(const uint4*)(Qb + bh + (size_t)(q0 + row) * HD + c8);
    }

    const int rowlo = (lane >> 4) << 3;
    const int col   = lane & 15;
    v8f   oacc[4] = {};
    float mrow[8], lrow[8];
#pragma unroll
    for (int r = 0; r < 8; ++r) { mrow[r] = -1e30f; lrow[r] = 0.0f; }

    int cur = 0;
#if USE_TDM
    if (wid == 0) {   // prologue: DMA tile 0 into buffer 0 (one wave drives the TDM)
        tdm_load_tile_2d_bf16(AT_KS_OFF(0), Kb + bh, HD, SS, HD, HD, 32);
        tdm_load_tile_2d_bf16(AT_VS_OFF(0), Vb + bh, HD, SS, HD, HD, 32);
    }
#endif

    for (int j0 = 0; j0 < SS; j0 += 32) {
#if USE_TDM
        if (wid == 0) {
            if (j0 + 32 < SS) {   // prefetch next tile into alternate buffer
                tdm_load_tile_2d_bf16(AT_KS_OFF(cur ^ 1), Kb + bh + (size_t)(j0 + 32) * HD,
                                      HD, SS, HD, HD, 32);
                tdm_load_tile_2d_bf16(AT_VS_OFF(cur ^ 1), Vb + bh + (size_t)(j0 + 32) * HD,
                                      HD, SS, HD, HD, 32);
                __builtin_amdgcn_s_wait_tensorcnt(2);  // current tile's 2 DMAs retired
            } else {
                __builtin_amdgcn_s_wait_tensorcnt(0);
            }
        }
#else
        {   // fallback: cooperative staging of current tile (row stride 66)
            __bf16* Kd = (__bf16*)(smem + AT_KS_OFF(cur));
            __bf16* Vd = (__bf16*)(smem + AT_VS_OFF(cur));
#pragma unroll
            for (int j = 0; j < 8; ++j) {
                int e2  = tid * 8 + j;          // 1024 pairs
                int row = e2 >> 5;
                int c2  = (e2 & 31) * 2;
                *(bf16x2*)(Kd + row * 66 + c2) = *(const bf16x2*)(Kb + bh + (size_t)(j0 + row) * HD + c2);
                *(bf16x2*)(Vd + row * 66 + c2) = *(const bf16x2*)(Vb + bh + (size_t)(j0 + row) * HD + c2);
            }
        }
#endif
        __syncthreads();

        const __bf16* Kc = (const __bf16*)(smem + AT_KS_OFF(cur));
        const __bf16* Vc = (const __bf16*)(smem + AT_VS_OFF(cur));

        // S = (Q/8) K^T : two 16x16 tiles, contraction over d=64 in two steps
        v8f s0 = {}, s1 = {};
#pragma unroll
        for (int kk = 0; kk < HD; kk += 32) {
            v16bf aq = load_frag_a(Qs + wid * 16 * 72 + kk, 72, lane);
            v16bf b0 = load_frag_b(Kc + kk, 66, lane, 0);
            v16bf b1 = load_frag_b(Kc + kk, 66, lane, 16);
            s0 = wmma_bf16(aq, b0, s0);
            s1 = wmma_bf16(aq, b1, s1);
        }

        // Online softmax (row = VGPR index, N striped over 16 lanes)
        __bf16* Pw = Pw0 + wid * 16 * 40;
        float alpha[8];
#pragma unroll
        for (int r = 0; r < 8; ++r) {
            float mx = fmaxf(s0[r], s1[r]);
#pragma unroll
            for (int off = 8; off >= 1; off >>= 1) mx = fmaxf(mx, __shfl_xor(mx, off, 16));
            float mnew = fmaxf(mrow[r], mx);
            alpha[r] = __expf(mrow[r] - mnew);
            mrow[r] = mnew;
            float p0 = __expf(s0[r] - mnew);
            float p1 = __expf(s1[r] - mnew);
            float rs = p0 + p1;
#pragma unroll
            for (int off = 8; off >= 1; off >>= 1) rs += __shfl_xor(rs, off, 16);
            lrow[r] = lrow[r] * alpha[r] + rs;
            Pw[(rowlo + r) * 40 + col]      = f2bf(p0);
            Pw[(rowlo + r) * 40 + 16 + col] = f2bf(p1);
        }
#pragma unroll
        for (int t = 0; t < 4; ++t)
#pragma unroll
            for (int r = 0; r < 8; ++r) oacc[t][r] *= alpha[r];

        // O += P @ V  (per-wave P tile; same-wave in-order DS pipeline)
        v16bf ap = load_frag_a(Pw, 40, lane);
#pragma unroll
        for (int t = 0; t < 4; ++t) {
            v16bf bv = load_frag_b_rowmajor(Vc, 66, lane, t * 16);
            oacc[t] = wmma_bf16(ap, bv, oacc[t]);
        }
        __syncthreads();   // all waves done with buf[cur] before TDM overwrites it
#if USE_TDM
        cur ^= 1;
#endif
    }

    // Normalize and write attn output as bf16 [b][s][h*64+d]
    float inv[8];
#pragma unroll
    for (int r = 0; r < 8; ++r) inv[r] = 1.0f / lrow[r];
#pragma unroll
    for (int t = 0; t < 4; ++t) {
#pragma unroll
        for (int r = 0; r < 8; ++r) {
            const int row  = wid * 16 + rowlo + r;
            const int colg = t * 16 + col;
            AOb[((size_t)bz * SS + q0 + row) * HH + h * HD + colg] = f2bf(oacc[t][r] * inv[r]);
        }
    }
}

// ---------------------------------------------------------------------------
// Kernel 3: out = attn_out(bf16) @ w_out + b_out -> fp32
// ---------------------------------------------------------------------------
__global__ __launch_bounds__(256) void out_gemm_kernel(
    const __bf16* __restrict__ A,    // [BS][HH] bf16
    const float*  __restrict__ W,    // [HH][HH]
    const float*  __restrict__ bias, // [HH]
    float* __restrict__ out)         // [BS][HH]
{
    __shared__ __bf16 As[64][40];
    __shared__ __bf16 BsT[128][40];

    const int tid  = threadIdx.x;
    const int lane = tid & 31;
    const int wid  = tid >> 5;
    const int wm   = wid & 3;
    const int wn   = wid >> 2;
    const int m0   = blockIdx.y * 64;
    const int n0   = blockIdx.x * 128;

    v8f acc[4] = {};

    for (int k0 = 0; k0 < HH; k0 += 32) {
        {   // Stage A: 64x32 bf16 (256 uint4, 1 per thread)
            int row = tid >> 2;
            int c8  = (tid & 3) * 8;
            *(uint4*)&As[row][c8] = *(const uint4*)(A + (size_t)(m0 + row) * HH + k0 + c8);
        }
        // Stage B transposed: 32x128 fp32 -> bf16
#pragma unroll
        for (int j = 0; j < 4; ++j) {
            int idx = tid * 4 + j;
            int row = idx >> 5;
            int c4  = (idx & 31) << 2;
            float4 f = *(const float4*)(W + (size_t)(k0 + row) * HH + n0 + c4);
            BsT[c4 + 0][row] = f2bf(f.x);
            BsT[c4 + 1][row] = f2bf(f.y);
            BsT[c4 + 2][row] = f2bf(f.z);
            BsT[c4 + 3][row] = f2bf(f.w);
        }
        if (k0 + 32 < HH) {
            __builtin_prefetch(A + (size_t)(m0 + (tid >> 2)) * HH + k0 + 32, 0, 1);
            __builtin_prefetch(W + (size_t)(k0 + 32 + (tid >> 3)) * HH + n0, 0, 1);
        }
        __syncthreads();

        v16bf a = load_frag_a(&As[wm * 16][0], 40, lane);
#pragma unroll
        for (int t = 0; t < 4; ++t) {
            v16bf b = load_frag_b(&BsT[0][0], 40, lane, wn * 64 + t * 16);
            acc[t] = wmma_bf16(a, b, acc[t]);
        }
        __syncthreads();
    }

    const int rowlo = (lane >> 4) << 3;
    const int col   = lane & 15;
#pragma unroll
    for (int t = 0; t < 4; ++t) {
        const int n  = n0 + wn * 64 + t * 16 + col;
        const float bv = bias[n];
#pragma unroll
        for (int r = 0; r < 8; ++r) {
            const int m = m0 + wm * 16 + rowlo + r;
            out[(size_t)m * HH + n] = acc[t][r] + bv;
        }
    }
}

// ---------------------------------------------------------------------------
extern "C" void kernel_launch(void* const* d_in, const int* in_sizes, int n_in,
                              void* d_out, int out_size, void* d_ws, size_t ws_size,
                              hipStream_t stream) {
    (void)in_sizes; (void)n_in; (void)out_size; (void)ws_size;
    const float* x     = (const float*)d_in[0];
    const float* w_qkv = (const float*)d_in[1];
    const float* b_qkv = (const float*)d_in[2];
    const float* w_out = (const float*)d_in[3];
    const float* b_out = (const float*)d_in[4];
    float* out = (float*)d_out;

    const size_t QKV_ELEMS = (size_t)BB * NH * SS * HD;   // 4 Mi elems
    __bf16* Qb  = (__bf16*)d_ws;
    __bf16* Kb  = Qb + QKV_ELEMS;
    __bf16* Vb  = Kb + QKV_ELEMS;
    __bf16* AOb = Vb + QKV_ELEMS;                         // [BS][HH] bf16

    qkv_gemm_kernel<<<dim3(N3 / 128, BS / 64), 256, 0, stream>>>(x, w_qkv, b_qkv, Qb, Kb, Vb);
    attn_kernel<<<dim3(SS / 64, NH, BB), 128, 0, stream>>>(Qb, Kb, Vb, AOb);
    out_gemm_kernel<<<dim3(HH / 128, BS / 64), 256, 0, stream>>>(AOb, w_out, b_out, out);
}
